// STFConvLayer_532575945001
// MI455X (gfx1250) — matrix operand order repaired and checked by reading, not compiled, not run on hardware
//
#include <hip/hip_runtime.h>

typedef __bf16 bf16_t;
typedef bf16_t v16bf __attribute__((ext_vector_type(16)));
typedef bf16_t v8bf  __attribute__((ext_vector_type(8)));
typedef bf16_t v2bf  __attribute__((ext_vector_type(2)));
typedef float  v8f   __attribute__((ext_vector_type(8)));
typedef float  v2f   __attribute__((ext_vector_type(2)));
typedef unsigned short u16;
typedef unsigned int   u32;

// ---------- helpers ----------
__device__ __forceinline__ u16 f2bf(float f) {   // init-time only
  union { float f; u32 u; } v; v.f = f;
  u32 r = v.u + 0x7FFFu + ((v.u >> 16) & 1u);    // round-to-nearest-even
  return (u16)(r >> 16);
}

// pack 2 f32 -> 2 bf16 in one v_cvt_pk_bf16_f32 (vector fptrunc pattern)
__device__ __forceinline__ u32 pk2(float a, float b) {
  v2f x; x[0] = a; x[1] = b;
  union { v2bf v; u32 u; } t;
  t.v = __builtin_convertvector(x, v2bf);
  return t.u;
}
// pack 8 f32 accumulator -> uint4 of bf16 (4x v_cvt_pk_bf16_f32)
__device__ __forceinline__ uint4 pk8(v8f a) {
  union { v8bf v; uint4 q; } t;
  t.v = __builtin_convertvector(a, v8bf);
  return t.q;
}

// A-matrix 16x32 bf16 fragment (ISA 7.12.2 "16-bit A-Matrix 16x32"):
// lane L: m = L&15, h = L>>4; element j: k = (j&8)*2 + h*8 + (j&7)
// -> two 16-byte chunks at k0 + h*8 and k0 + 16 + h*8 of row m.
__device__ __forceinline__ v16bf frag_a(const u16* rowk, int h) {
  union { uint4 u[2]; v16bf v; } t;
  t.u[0] = *(const uint4*)(rowk + h * 8);
  t.u[1] = *(const uint4*)(rowk + 16 + h * 8);
  return t.v;
}
// B-matrix 32x16 bf16 fragment (same k order as A):
// lane L: n = L&15, h = L>>4; element j: k = (j&8)*2 + h*8 + (j&7).
// Source stored as B^T rows (row n, K contiguous) -> two 16-byte chunks.
__device__ __forceinline__ v16bf frag_b(const u16* rowk, int h) {
  union { uint4 u[2]; v16bf v; } t;
  t.u[0] = *(const uint4*)(rowk + h * 8);
  t.u[1] = *(const uint4*)(rowk + 16 + h * 8);
  return t.v;
}

__device__ __forceinline__ v8f wmma_bf16(v16bf a, v16bf b, v8f c) {
  return __builtin_amdgcn_wmma_f32_16x16x32_bf16(false, a, false, b, (short)0, c,
                                                 false, false);
}

// ---------- workspace layout (bf16 elements) ----------
// W2T [64][384]   @ 0       conv weights transposed, K=(p,t,c)
// F16 [32][32]    @ 24576   fwd DFT: rows [0,16)=real bins (0-pad), [16,32)=imag
// F32 [64][32]    @ 25600   rows [0,32)=real, [32,64)=imag
// F64 [96][64]    @ 27648   rows [0,48)=real, [48,96)=imag
// C16 [16][32]    @ 33792   inv DFT; cols [0,bins)=real, [binsp,binsp+bins)=imag
// C32 [32][64]    @ 34304   (binsp = 16/24/40, gap + tail columns are zero)
// C64 [64][96]    @ 36352   ; total 42496 elems = 85 KB
__global__ __launch_bounds__(256) void init_ws_kernel(const float* __restrict__ kr,
                                                      const float* __restrict__ ki,
                                                      u16* __restrict__ ws) {
  int idx = blockIdx.x * 256 + threadIdx.x;
  if (idx >= 42496) return;
  const float TWO_PI = 6.283185307179586f;
  float val = 0.0f;
  if (idx < 24576) {
    // W2T[n][kappa], kappa = p*192 + t*64 + c
    int n = idx / 384, kap = idx - n * 384;
    int p = kap / 192, r = kap - p * 192;
    int t = r / 64, c = r - t * 64;
    int o = n & 31;
    float wr = kr[(t * 64 + c) * 32 + o];
    float wi = ki[(t * 64 + c) * 32 + o];
    // out_r = rp*Wr - ip*Wi ; out_i = rp*Wi + ip*Wr
    val = (n < 32) ? (p ? -wi : wr) : (p ? wr : wi);
  } else {
    int i = idx - 24576;
    int fftn, cols, aux, kind;   // aux: kind0 -> M2P, kind1 -> BINSP
    if (i < 1024)       {             fftn = 16; cols = 32; aux = 16; kind = 0; }
    else if (i < 3072)  { i -= 1024;  fftn = 32; cols = 32; aux = 32; kind = 0; }
    else if (i < 9216)  { i -= 3072;  fftn = 64; cols = 64; aux = 48; kind = 0; }
    else if (i < 9728)  { i -= 9216;  fftn = 16; cols = 32; aux = 16; kind = 1; }
    else if (i < 11776) { i -= 9728;  fftn = 32; cols = 64; aux = 24; kind = 1; }
    else                { i -= 11776; fftn = 64; cols = 96; aux = 40; kind = 1; }
    int m = i / cols, nk = i - m * cols;
    int bins = fftn / 2 + 1;
    if (kind == 0) {               // forward DFT rows (real block | imag block)
      int p  = (m >= aux) ? 1 : 0;
      int kb = m - p * aux;
      if (kb < bins && nk < fftn)
        val = p ? -sinf(TWO_PI * kb * nk / fftn) : cosf(TWO_PI * kb * nk / fftn);
    } else {                       // inverse DFT, row m = time index
      if (nk < bins) {
        int k = nk;
        float w = (k == 0 || k == bins - 1) ? 1.0f : 2.0f;
        val = (w / fftn) * cosf(TWO_PI * k * m / fftn);
      } else if (nk >= aux && nk < aux + bins) {
        int k = nk - aux;
        val = (k == 0 || k == bins - 1) ? 0.0f
                                        : (-2.0f / fftn) * sinf(TWO_PI * k * m / fftn);
      }
    }
  }
  ws[idx] = f2bf(val);
}

// ---------- fused per-frame kernel: 1 wave per frame ----------
template <int FFTN>
__global__ __launch_bounds__(32) void stft_conv_kernel(
    const float* __restrict__ x, const u16* __restrict__ ws,
    float* __restrict__ out, int chOff, int fOff, int cOff) {
  constexpr int BINS  = FFTN / 2 + 1;
  constexpr int BINSP = (BINS + 7) & ~7;                  // 16-B aligned imag base
  constexpr int DIL   = FFTN / 16;
  constexpr int FRAMES = 4096 / FFTN;
  constexpr int K1    = (FFTN < 32) ? 32 : FFTN;          // stage1 K (padded)
  constexpr int M2T   = (BINS + 15) / 16;                 // stage2 M tiles
  constexpr int M2P   = M2T * 16;
  constexpr int M1T   = 2 * M2T;                          // stage1 spec tiles (re|im)
  constexpr int ROWSP = M2P + 2 * DIL;                    // padded spec rows per p
  constexpr int K3    = ((BINSP + BINS + 31) / 32) * 32;  // stage3 K (padded)
  constexpr int M3T   = FFTN / 16;
  // LDS layout (u16 elements)
  constexpr int XFT = 0;                     // XfT[64][K1]
  constexpr int SP  = 64 * K1;               // S_pad[2][ROWSP][64]
  constexpr int OT  = SP + 2 * ROWSP * 64;   // O'T[32][K3]
  constexpr int LDSE = OT + 32 * K3;         // total elems

  extern __shared__ u16 lds[];
  bf16_t* ldsb = (bf16_t*)lds;

  const int lane = threadIdx.x;
  const int lm  = lane & 15;
  const int h   = lane >> 4;
  const int hi8 = h * 8;

  const int fi = blockIdx.x;
  const int b = fi / FRAMES;
  const int f = fi - b * FRAMES;

  const float* __restrict__ xf = x + ((size_t)b * 4096 + (size_t)f * FFTN) * 64;
  const u16* __restrict__ Fm = ws + fOff;
  const u16* __restrict__ Ci = ws + cOff;
  const u16* __restrict__ W2 = ws;

  // zero entire slice once: guarantees exact K-padding zeros and finite values
  {
    uint4 z; z.x = z.y = z.z = z.w = 0u;
    uint4* l4 = (uint4*)lds;
    for (int i = lane; i < LDSE / 8; i += 32) l4[i] = z;
  }

  // load frame, transpose to XfT[c][n]; two time samples packed per b32 store
  for (int i = lane; i < FFTN * 8; i += 32) {
    int n  = (i >> 4) * 2;
    int c4 = (i & 15) * 4;
    float4 va = *(const float4*)(xf + n * 64 + c4);
    float4 vb = *(const float4*)(xf + (n + 1) * 64 + c4);
    *(u32*)(ldsb + XFT + (c4 + 0) * K1 + n) = pk2(va.x, vb.x);
    *(u32*)(ldsb + XFT + (c4 + 1) * K1 + n) = pk2(va.y, vb.y);
    *(u32*)(ldsb + XFT + (c4 + 2) * K1 + n) = pk2(va.z, vb.z);
    *(u32*)(ldsb + XFT + (c4 + 3) * K1 + n) = pk2(va.w, vb.w);
  }

  // ---- Stage 1: rFFT as GEMM, computed transposed:
  // D[c][specrow] = sum_k XfT[c][k] * F[specrow][k]  (A = XfT, B^T rows = F rows)
  // Lane's 8 outputs = 8 consecutive channels at one spec row -> 1 b128 store.
  #pragma unroll
  for (int mtc = 0; mtc < 4; ++mtc) {            // channel tiles (M = 64)
    #pragma unroll
    for (int ntf = 0; ntf < M1T; ++ntf) {        // spec-row tiles (N)
      v8f acc = {};
      #pragma unroll
      for (int kc = 0; kc < K1 / 32; ++kc) {
        v16bf a  = frag_a(lds + XFT + (mtc * 16 + lm) * K1 + kc * 32, h);
        v16bf bm = frag_b(Fm + (ntf * 16 + lm) * K1 + kc * 32, h);
        acc = wmma_bf16(a, bm, acc);
      }
      const int p   = ntf / M2T;                 // constants after unroll
      const int kb  = (ntf - p * M2T) * 16 + lm; // spec row (lane col)
      *(uint4*)(ldsb + SP + (p * ROWSP + kb + DIL) * 64 + mtc * 16 + hi8) = pk8(acc);
    }
  }

  // ---- Hermitian reflect-pad (imag rows sign-flipped) ----
  for (int i = lane; i < 4 * DIL * 64; i += 32) {
    int c  = i & 63;
    int r2 = i >> 6;
    int p  = r2 & 1;
    int ms = r2 >> 1;                          // 0..2*DIL-1
    int side = (ms >= DIL) ? 1 : 0;
    int m = (side ? ms - DIL : ms) + 1;        // 1..DIL
    int dst = side ? (DIL + BINS - 1 + m) : (DIL - m);
    int src = side ? (DIL + BINS - 1 - m) : (DIL + m);
    u16 v = lds[SP + (p * ROWSP + src) * 64 + c];
    if (p) v ^= 0x8000u;                       // negate bf16
    lds[SP + (p * ROWSP + dst) * 64 + c] = v;
  }

  // ---- Stage 2: complex dilated conv as GEMM (K=384), LeakyReLU -> O'T ----
  // nt OUTER: all real-channel writes precede imag writes, so real-block
  // overflow columns are later overwritten by valid imag data; remaining
  // overflow lands in columns where Cinv is exactly 0.0 (finite garbage ok).
  #pragma unroll
  for (int nt = 0; nt < 4; ++nt) {
    const int n    = nt * 16 + lm;
    const int o    = n & 31;
    const int col0 = (nt >= 2) ? BINSP : 0;
    #pragma unroll
    for (int mt = 0; mt < M2T; ++mt) {
      v8f acc = {};
      #pragma unroll
      for (int kk = 0; kk < 12; ++kk) {        // kappa0 = p*192 + t*64 + ch*32
        int p  = kk / 6;
        int t  = (kk % 6) / 2;
        int ch = kk & 1;
        v16bf a  = frag_a(lds + SP + (p * ROWSP + mt * 16 + lm + t * DIL) * 64 + ch * 32, h);
        v16bf bm = frag_b(W2 + n * 384 + p * 192 + t * 64 + ch * 32, h);
        acc = wmma_bf16(a, bm, acc);
      }
      #pragma unroll
      for (int v = 0; v < 8; ++v)
        acc[v] = fmaxf(acc[v], 0.2f * acc[v]);    // leaky_relu (mul+max)
      *(uint4*)(ldsb + OT + o * K3 + col0 + mt * 16 + hi8) = pk8(acc);
    }
  }

  // ---- Stage 3: irFFT as GEMM, computed transposed:
  // D[o][ntime] = sum_k O'T[o][k] * Cinv[ntime][k]  (A = O'T, B^T rows = Cinv)
  // Lane's 8 outputs = 8 consecutive out channels at one time -> 2 b128 stores.
  #pragma unroll
  for (int mt2 = 0; mt2 < 2; ++mt2) {            // channel tiles (M = 32)
    #pragma unroll
    for (int ntf = 0; ntf < M3T; ++ntf) {        // time tiles (N)
      v8f acc = {};
      #pragma unroll
      for (int kc = 0; kc < K3 / 32; ++kc) {
        v16bf a  = frag_a(lds + OT + (mt2 * 16 + lm) * K3 + kc * 32, h);
        v16bf bm = frag_b(Ci + (ntf * 16 + lm) * K3 + kc * 32, h);
        acc = wmma_bf16(a, bm, acc);
      }
      const int ntime = ntf * 16 + lm;           // lane col = time index
      float* orow = out + ((size_t)b * 4096 + (size_t)f * FFTN + ntime) * 96
                        + chOff + mt2 * 16 + hi8;
      union { v8f v; float4 f4[2]; } st; st.v = acc;
      ((float4*)orow)[0] = st.f4[0];
      ((float4*)orow)[1] = st.f4[1];
    }
  }
}

// ---------- launcher ----------
extern "C" void kernel_launch(void* const* d_in, const int* in_sizes, int n_in,
                              void* d_out, int out_size, void* d_ws, size_t ws_size,
                              hipStream_t stream) {
  (void)in_sizes; (void)n_in; (void)out_size; (void)ws_size;
  const float* x  = (const float*)d_in[0];
  const float* kr = (const float*)d_in[1];
  const float* ki = (const float*)d_in[2];
  float* out = (float*)d_out;
  u16* ws = (u16*)d_ws;

  // build bf16 DFT / iDFT / weight matrices in workspace (85 KB)
  init_ws_kernel<<<166, 256, 0, stream>>>(kr, ki, ws);

  // one wave (32 threads) per frame; dynamic LDS sized per branch
  stft_conv_kernel<16><<<16 * 256, 32, 10752, stream>>>(x, ws, out, 0,  24576, 33792);
  stft_conv_kernel<32><<<16 * 128, 32, 17408, stream>>>(x, ws, out, 32, 25600, 34304);
  stft_conv_kernel<64><<<16 * 64,  32, 28672, stream>>>(x, ws, out, 64, 27648, 36352);
}
